// ContextCluster_69612829933695
// MI455X (gfx1250) — matrix operand order repaired and checked by reading, not compiled, not run on hardware
//
#include <hip/hip_runtime.h>
#include <hip/hip_bf16.h>
#include <math.h>

typedef __attribute__((ext_vector_type(16))) __bf16 v16bf;
typedef __attribute__((ext_vector_type(8)))  float  v8f;

union U16 { uint4 u2[2]; v16bf v; unsigned short s[16]; };
union U8  { uint4 u;     unsigned short s[8]; unsigned w[4]; };

__device__ __forceinline__ unsigned short f2bf(float f) {
    unsigned u = __float_as_uint(f);
    u += 0x7FFFu + ((u >> 16) & 1u);          // round-to-nearest-even
    return (unsigned short)(u >> 16);
}
__device__ __forceinline__ float bf2f(unsigned short s) {
    return __uint_as_float(((unsigned)s) << 16);
}
// packed f32x2 -> bf16x2 : single v_cvt_pk_bf16_f32 (CDNA bf16 convert pipe)
__device__ __forceinline__ unsigned pk2bf(float a, float b) {
#if __has_builtin(__builtin_amdgcn_cvt_pk_bf16_f32)
    auto p = __builtin_amdgcn_cvt_pk_bf16_f32(a, b);
    return __builtin_bit_cast(unsigned, p);
#else
    unsigned r;
    asm("v_cvt_pk_bf16_f32 %0, %1, %2" : "=v"(r) : "v"(a), "v"(b));
    return r;
#endif
}

#define WMMA_BF16(a, b, c) \
    __builtin_amdgcn_wmma_f32_16x16x32_bf16(false, (a), false, (b), (short)0, (c), false, false)

// ---- CDNA5 async global->LDS copy (ASYNCcnt path, no VGPR data movement)
__device__ __forceinline__ void async_g2l_b128(void* lds_dst, const void* gsrc) {
    unsigned loff = (unsigned)(unsigned long long)lds_dst;   // low 32 bits = LDS offset
    asm volatile("global_load_async_to_lds_b128 %0, %1, off"
                 :: "v"(loff), "v"(gsrc) : "memory");
}
__device__ __forceinline__ void wait_async0() {
    asm volatile("s_wait_asynccnt 0" ::: "memory");
}

// ---------------------------------------------------------------- weights f32->bf16 (paired)
__global__ __launch_bounds__(256) void k_cvt(const float* __restrict__ s,
                                             unsigned short* __restrict__ d, int n) {
    int i = (blockIdx.x * 256 + threadIdx.x) * 2;
    if (i < n) {
        float2 s2 = *(const float2*)&s[i];
        *(unsigned*)&d[i] = pk2bf(s2.x, s2.y);
    }
}

// ---------------------------------------------------------------- fused f/v GEMM
// rows = b*3136 + p  (p innermost over 56x56), cols 0..383 = f, 384..767 = v
// block tile: 64 positions x 128 outputs, 8 waves as 2(m) x 4(n), 2x2 tiles/wave
__global__ __launch_bounds__(256) void k_fv(const float* __restrict__ x,
                                            const unsigned short* __restrict__ Wfb,
                                            const unsigned short* __restrict__ Wvb,
                                            const float* __restrict__ bf_,
                                            const float* __restrict__ bv_,
                                            unsigned short* __restrict__ fv) {
    __shared__ unsigned short lds_a[64 * 32];   // [pos][ch] bf16
    const int tid  = threadIdx.x;
    const int lane = tid & 31;
    const int wave = tid >> 5;
    const int b    = (blockIdx.x * 64) / 3136;
    const int p0   = (blockIdx.x * 64) % 3136;
    const int wm   = wave >> 2, wn = wave & 3;
    const int d0   = blockIdx.y * 128 + wn * 32;     // y<3 -> f, y>=3 -> v (tile never crosses)
    const bool isV = d0 >= 384;
    const unsigned short* Wsel = isV ? Wvb : Wfb;
    const float*          bias = isV ? bv_ : bf_;
    const int dbase = isV ? d0 - 384 : d0;
    const int n16 = lane & 15, hi = lane >> 4;
    const size_t xbase = (size_t)(b * 384) * 3136 + p0;

    v8f acc[2][2] = {};

    for (int k0 = 0; k0 < 384; k0 += 32) {
        __syncthreads();
        {   // stage A tile: 32 channels x 64 positions -> lds [pos][ch] bf16
            int cl = tid >> 3;            // channel 0..31
            int pj = (tid & 7) << 3;      // position 0..56 step 8
            const float* xs = x + xbase + (size_t)(k0 + cl) * 3136 + pj;
            float4 x0 = *(const float4*)xs;
            float4 x1 = *(const float4*)(xs + 4);
            unsigned q01 = pk2bf(x0.x, x0.y), q23 = pk2bf(x0.z, x0.w);
            unsigned q45 = pk2bf(x1.x, x1.y), q67 = pk2bf(x1.z, x1.w);
            lds_a[(pj + 0) * 32 + cl] = (unsigned short)q01;
            lds_a[(pj + 1) * 32 + cl] = (unsigned short)(q01 >> 16);
            lds_a[(pj + 2) * 32 + cl] = (unsigned short)q23;
            lds_a[(pj + 3) * 32 + cl] = (unsigned short)(q23 >> 16);
            lds_a[(pj + 4) * 32 + cl] = (unsigned short)q45;
            lds_a[(pj + 5) * 32 + cl] = (unsigned short)(q45 >> 16);
            lds_a[(pj + 6) * 32 + cl] = (unsigned short)q67;
            lds_a[(pj + 7) * 32 + cl] = (unsigned short)(q67 >> 16);
            if (k0 + 32 < 384)
                __builtin_prefetch(xs + (size_t)32 * 3136, 0, 1);
        }
        __syncthreads();
        U16 A0, A1, B0, B1;
        const int ar0 = (wm * 32 + n16) * 32;
        const int ar1 = (wm * 32 + 16 + n16) * 32;
        A0.u2[0] = *(const uint4*)&lds_a[ar0 + hi * 16];
        A0.u2[1] = *(const uint4*)&lds_a[ar0 + hi * 16 + 8];
        A1.u2[0] = *(const uint4*)&lds_a[ar1 + hi * 16];
        A1.u2[1] = *(const uint4*)&lds_a[ar1 + hi * 16 + 8];
        const int kb = k0 + hi * 16;
        const unsigned short* w0 = &Wsel[(dbase + n16) * 384 + kb];
        const unsigned short* w1 = &Wsel[(dbase + 16 + n16) * 384 + kb];
        B0.u2[0] = *(const uint4*)w0;  B0.u2[1] = *(const uint4*)(w0 + 8);
        B1.u2[0] = *(const uint4*)w1;  B1.u2[1] = *(const uint4*)(w1 + 8);
        acc[0][0] = WMMA_BF16(A0.v, B0.v, acc[0][0]);
        acc[0][1] = WMMA_BF16(A0.v, B1.v, acc[0][1]);
        acc[1][0] = WMMA_BF16(A1.v, B0.v, acc[1][0]);
        acc[1][1] = WMMA_BF16(A1.v, B1.v, acc[1][1]);
    }
#pragma unroll
    for (int im = 0; im < 2; im++)
#pragma unroll
        for (int jn = 0; jn < 2; jn++) {
            const float bb  = bias[dbase + jn * 16 + n16];
            const int   col = d0 + jn * 16 + n16;
            const size_t base = (size_t)(b * 3136 + p0 + wm * 32 + im * 16 + hi * 8) * 768 + col;
#pragma unroll
            for (int i = 0; i < 8; i += 2) {
                unsigned q = pk2bf(acc[im][jn][i] + bb, acc[im][jn][i + 1] + bb);
                fv[base + (size_t)i * 768]       = (unsigned short)q;
                fv[base + (size_t)(i + 1) * 768] = (unsigned short)(q >> 16);
            }
        }
}

// ---------------------------------------------------------------- clustering (1 block / group)
#define FPAD_OFF 0          // bf16 [784][64]  (K zero-padded 48->64)
#define VT_OFF   100352     // bf16 [48][800]  (tokens zero-padded 784->800)
#define SIMF_OFF 177152     // f32  [16][788]
#define SIMM_OFF 227584     // bf16 [16][800]
#define CEN_OFF  253184     // bf16 [16][64]
#define VC_OFF   255232     // f32  [16][48]
#define AGG_OFF  258304     // f32  [16][48]
#define INVT_OFF 261376     // f32  [784]
#define IDX_OFF  264512     // int  [784]
#define INVC_OFF 267648     // f32  [16]
#define DEN_OFF  267712     // f32  [16]
#define CL_LDS   267776

__global__ __launch_bounds__(256) void k_cluster(const unsigned short* __restrict__ fv,
                                                 unsigned short* __restrict__ oarr,
                                                 const float* __restrict__ alpha_p,
                                                 const float* __restrict__ beta_p) {
    extern __shared__ char smem[];
    unsigned short* FP   = (unsigned short*)(smem + FPAD_OFF);
    unsigned short* VT   = (unsigned short*)(smem + VT_OFF);
    float*          SIMF = (float*)(smem + SIMF_OFF);
    unsigned short* SIMM = (unsigned short*)(smem + SIMM_OFF);
    unsigned short* CEN  = (unsigned short*)(smem + CEN_OFF);
    float*          VC   = (float*)(smem + VC_OFF);
    float*          AGG  = (float*)(smem + AGG_OFF);
    float*          INVT = (float*)(smem + INVT_OFF);
    int*            IDX  = (int*)(smem + IDX_OFF);
    float*          INVC = (float*)(smem + INVC_OFF);
    float*          DEN  = (float*)(smem + DEN_OFF);

    const int tid = threadIdx.x;
    const int g = blockIdx.x;
    const int f2 = g & 1, f1 = (g >> 1) & 1, e = (g >> 2) & 7, b = g >> 5;
    const int base_row = b * 3136 + f1 * 28 * 56 + f2 * 28;
    const int fcol = e * 48;
    const int vcol = 384 + e * 48;
    const float alpha = *alpha_p, beta = *beta_p;

    if (tid < 16) { INVC[tid] = 0.f; DEN[tid] = 0.f; }

    // ---- load f -> FP [t][64] via async DMA (pad K 48..63 with zeros)
    for (int it = tid; it < 784 * 8; it += 256) {
        int t = it >> 3, j = it & 7;
        int w = t / 28, h = t - w * 28;
        int row = base_row + w * 56 + h;
        if (j < 6) {
            async_g2l_b128(&FP[t * 64 + j * 8], &fv[(size_t)row * 768 + fcol + j * 8]);
        } else {
            uint4 z = {0, 0, 0, 0};
            *(uint4*)&FP[t * 64 + j * 8] = z;
        }
    }
    // ---- v -> VT [cc][800] (transposed through VGPRs)
    for (int it = tid; it < 784 * 6; it += 256) {
        int t = it / 6, j = it - t * 6;
        int w = t / 28, h = t - w * 28;
        int row = base_row + w * 56 + h;
        U8 d; d.u = *(const uint4*)&fv[(size_t)row * 768 + vcol + j * 8];
#pragma unroll
        for (int q = 0; q < 8; q++) VT[(j * 8 + q) * 800 + t] = d.s[q];
    }
    for (int it = tid; it < 48 * 16; it += 256)
        VT[(it >> 4) * 800 + 784 + (it & 15)] = 0;
    wait_async0();
    __syncthreads();

    // ---- centers, pooled v, norms
    for (int k = tid; k < 768; k += 256) {
        int m = k / 48, cc = k - m * 48;
        int pw = m >> 2, ph = m & 3;
        float sf = 0.f, sv = 0.f;
        for (int dw = 0; dw < 7; dw++)
            for (int dh = 0; dh < 7; dh++) {
                int t = (pw * 7 + dw) * 28 + (ph * 7 + dh);
                sf += bf2f(FP[t * 64 + cc]);
                sv += bf2f(VT[cc * 800 + t]);
            }
        float cf = sf * (1.f / 49.f);
        CEN[m * 64 + cc] = f2bf(cf);
        VC[m * 48 + cc]  = sv * (1.f / 49.f);
        atomicAdd(&INVC[m], cf * cf);
    }
    for (int k = tid; k < 256; k += 256)        // zero-pad center K 48..63
        CEN[(k >> 4) * 64 + 48 + (k & 15)] = 0;
    for (int t = tid; t < 784; t += 256) {
        float s = 0.f;
        for (int cc = 0; cc < 48; cc++) { float fz = bf2f(FP[t * 64 + cc]); s += fz * fz; }
        INVT[t] = 1.f / fmaxf(sqrtf(s), 1e-12f);
    }
    __syncthreads();
    if (tid < 16) INVC[tid] = 1.f / fmaxf(sqrtf(INVC[tid]), 1e-12f);
    __syncthreads();

    const int lane = tid & 31;
    const int wave = tid >> 5;
    const int n16 = lane & 15, hi = lane >> 4;

    // ---- sim = sigmoid(beta + alpha * cos(cen, tok)) via WMMA (K = 64 padded)
    U16 A0, A1;
    A0.u2[0] = *(const uint4*)&CEN[n16 * 64 + hi * 16];
    A0.u2[1] = *(const uint4*)&CEN[n16 * 64 + hi * 16 + 8];
    A1.u2[0] = *(const uint4*)&CEN[n16 * 64 + 32 + hi * 16];
    A1.u2[1] = *(const uint4*)&CEN[n16 * 64 + 32 + hi * 16 + 8];
    for (int nt = wave; nt < 49; nt += 8) {
        int t = nt * 16 + n16;
        U16 B0, B1;
        B0.u2[0] = *(const uint4*)&FP[t * 64 + hi * 16];
        B0.u2[1] = *(const uint4*)&FP[t * 64 + hi * 16 + 8];
        B1.u2[0] = *(const uint4*)&FP[t * 64 + 32 + hi * 16];
        B1.u2[1] = *(const uint4*)&FP[t * 64 + 32 + hi * 16 + 8];
        v8f c = {};
        c = WMMA_BF16(A0.v, B0.v, c);
        c = WMMA_BF16(A1.v, B1.v, c);
        float it_ = INVT[t];
#pragma unroll
        for (int i = 0; i < 8; i++) {
            int m = i + hi * 8;
            float cs = c[i] * it_ * INVC[m];
            SIMF[m * 788 + t] = 1.f / (1.f + __expf(-(beta + alpha * cs)));
        }
    }
    __syncthreads();

    // ---- hard assignment (first max), masked bf16 sim, denominators
    for (int t = tid; t < 784; t += 256) {
        float best = -1.f; int bi = 0;
#pragma unroll
        for (int m = 0; m < 16; m++) {
            float s = SIMF[m * 788 + t];
            if (s > best) { best = s; bi = m; }
        }
        IDX[t] = bi;
        unsigned short bb = f2bf(best);
#pragma unroll
        for (int m = 0; m < 16; m++)
            SIMM[m * 800 + t] = (m == bi) ? bb : (unsigned short)0;
        atomicAdd(&DEN[bi], best);
    }
    for (int k = tid; k < 256; k += 256)        // zero-pad masked sim tokens 784..799
        SIMM[(k >> 4) * 800 + 784 + (k & 15)] = 0;
    __syncthreads();

    // ---- agg[m][cc] = (sum_n simm*v + vc) / (den + 1) via WMMA (K = 800 tokens)
    if (wave < 3) {
        int cc = wave * 16 + n16;
        v8f c = {};
        for (int kt = 0; kt < 25; kt++) {
            int ko = kt * 32;
            U16 A, B;
            A.u2[0] = *(const uint4*)&SIMM[n16 * 800 + ko + hi * 16];
            A.u2[1] = *(const uint4*)&SIMM[n16 * 800 + ko + hi * 16 + 8];
            B.u2[0] = *(const uint4*)&VT[cc * 800 + ko + hi * 16];
            B.u2[1] = *(const uint4*)&VT[cc * 800 + ko + hi * 16 + 8];
            c = WMMA_BF16(A.v, B.v, c);
        }
#pragma unroll
        for (int i = 0; i < 8; i++) {
            int m = i + hi * 8;
            AGG[m * 48 + cc] = (c[i] + VC[m * 48 + cc]) / (DEN[m] + 1.f);
        }
    }
    __syncthreads();

    // ---- dispatch: out[t][cc] = sim[idx][t] * agg[idx][cc]
    for (int t = tid; t < 784; t += 256) {
        int id = IDX[t];
        float s = SIMF[id * 788 + t];
        int w = t / 28, h = t - w * 28;
        size_t row = (size_t)(base_row + w * 56 + h);
        unsigned short* dst = &oarr[row * 384 + fcol];
#pragma unroll
        for (int j = 0; j < 6; j++) {
            U8 o;
#pragma unroll
            for (int q = 0; q < 4; q++)
                o.w[q] = pk2bf(s * AGG[id * 48 + j * 8 + q * 2],
                               s * AGG[id * 48 + j * 8 + q * 2 + 1]);
            *(uint4*)&dst[j * 8] = o.u;
        }
    }
}

// ---------------------------------------------------------------- final projection GEMM
// block tile: 64 positions x 128 outputs; double-buffered async global->LDS A staging
__global__ __launch_bounds__(256) void k_proj(const unsigned short* __restrict__ ta,
                                              const unsigned short* __restrict__ Wpb,
                                              const float* __restrict__ bp_,
                                              float* __restrict__ y) {
    __shared__ unsigned short lds_a[2][64 * 32];
    const int tid  = threadIdx.x;
    const int lane = tid & 31;
    const int wave = tid >> 5;
    const int b  = (blockIdx.x * 64) / 3136;
    const int p0 = (blockIdx.x * 64) % 3136;
    const int wm = wave >> 2, wn = wave & 3;
    const int o0 = blockIdx.y * 128 + wn * 32;
    const int n16 = lane & 15, hi = lane >> 4;
    v8f acc[2][2] = {};

    const int pos = tid >> 2;             // 0..63
    const int ch8 = (tid & 3) << 3;       // 0,8,16,24
    const unsigned short* src_row = ta + (size_t)(b * 3136 + p0 + pos) * 384 + ch8;
    const unsigned stage_off = pos * 32 + ch8;

    async_g2l_b128(&lds_a[0][stage_off], src_row);      // prime buffer 0
    int cur = 0;

    for (int k0 = 0; k0 < 384; k0 += 32) {
        wait_async0();
        __syncthreads();                                 // lds_a[cur] ready for all waves
        if (k0 + 32 < 384)
            async_g2l_b128(&lds_a[cur ^ 1][stage_off], src_row + k0 + 32);   // overlap DMA
        const unsigned short* la = lds_a[cur];
        U16 A0, A1, B0, B1;
        const int ar0 = (wm * 32 + n16) * 32;
        const int ar1 = (wm * 32 + 16 + n16) * 32;
        A0.u2[0] = *(const uint4*)&la[ar0 + hi * 16];
        A0.u2[1] = *(const uint4*)&la[ar0 + hi * 16 + 8];
        A1.u2[0] = *(const uint4*)&la[ar1 + hi * 16];
        A1.u2[1] = *(const uint4*)&la[ar1 + hi * 16 + 8];
        const int kb = k0 + hi * 16;
        const unsigned short* w0 = &Wpb[(o0 + n16) * 384 + kb];
        const unsigned short* w1 = &Wpb[(o0 + 16 + n16) * 384 + kb];
        B0.u2[0] = *(const uint4*)w0;  B0.u2[1] = *(const uint4*)(w0 + 8);
        B1.u2[0] = *(const uint4*)w1;  B1.u2[1] = *(const uint4*)(w1 + 8);
        acc[0][0] = WMMA_BF16(A0.v, B0.v, acc[0][0]);
        acc[0][1] = WMMA_BF16(A0.v, B1.v, acc[0][1]);
        acc[1][0] = WMMA_BF16(A1.v, B0.v, acc[1][0]);
        acc[1][1] = WMMA_BF16(A1.v, B1.v, acc[1][1]);
        cur ^= 1;
    }
#pragma unroll
    for (int im = 0; im < 2; im++)
#pragma unroll
        for (int jn = 0; jn < 2; jn++) {
            const int o = o0 + jn * 16 + n16;
            const float bo = bp_[o];
            float* dst = y + ((size_t)b * 384 + o) * 3136 + p0 + wm * 32 + im * 16 + hi * 8;
            v8f a = acc[im][jn];
            float4 s0 = {a[0] + bo, a[1] + bo, a[2] + bo, a[3] + bo};
            float4 s1 = {a[4] + bo, a[5] + bo, a[6] + bo, a[7] + bo};
            *(float4*)dst = s0;
            *(float4*)(dst + 4) = s1;
        }
}

// ---------------------------------------------------------------- host
extern "C" void kernel_launch(void* const* d_in, const int* in_sizes, int n_in,
                              void* d_out, int out_size, void* d_ws, size_t ws_size,
                              hipStream_t stream) {
    const float* x     = (const float*)d_in[0];
    const float* Wf    = (const float*)d_in[1];
    const float* bf_   = (const float*)d_in[2];
    const float* Wv    = (const float*)d_in[3];
    const float* bv_   = (const float*)d_in[4];
    const float* Wp    = (const float*)d_in[5];
    const float* bp_   = (const float*)d_in[6];
    const float* alpha = (const float*)d_in[7];
    const float* beta  = (const float*)d_in[8];

    unsigned short* Wfb  = (unsigned short*)d_ws;
    unsigned short* Wvb  = Wfb + 147456;
    unsigned short* Wpb  = Wvb + 147456;
    unsigned short* fv   = Wpb + 147456;                       // [100352][768] bf16
    unsigned short* oarr = fv + (size_t)100352 * 768;          // [100352][384] bf16

    k_cvt<<<288, 256, 0, stream>>>(Wf, Wfb, 147456);
    k_cvt<<<288, 256, 0, stream>>>(Wv, Wvb, 147456);
    k_cvt<<<288, 256, 0, stream>>>(Wp, Wpb, 147456);

    dim3 gfv(1568, 6);
    k_fv<<<gfv, 256, 0, stream>>>(x, Wfb, Wvb, bf_, bv_, fv);

    (void)hipFuncSetAttribute((const void*)k_cluster,
                              hipFuncAttributeMaxDynamicSharedMemorySize, CL_LDS);
    k_cluster<<<1024, 256, CL_LDS, stream>>>(fv, oarr, alpha, beta);

    dim3 gp(1568, 3);
    k_proj<<<gp, 256, 0, stream>>>(oarr, Wpb, bp_, (float*)d_out);
}
